// HUH_83004537962778
// MI455X (gfx1250) — matrix operand order repaired and checked by reading, not compiled, not run on hardware
//
#include <hip/hip_runtime.h>

// Problem constants (from reference)
#define BB   64
#define TT   90
#define OO   5
#define DD   256
#define HH   256
#define EA   4096
#define RR   5
#define NB   8
#define NPG  (TT + OO)          // 95
#define NN   (BB * NPG)         // 6080
#define EPB  (EA + TT * OO)     // 4546 edges per batch
#define ETOT (BB * EPB)         // 290944 total edges
#define MSRC (BB * TT)          // 5760 source rows for Y

// gfx1250 async global->LDS path (tracked on ASYNCcnt), guarded per-toolchain
#if defined(__gfx1250__) && defined(__has_builtin)
#if __has_builtin(__builtin_amdgcn_global_load_async_to_lds_b128)
#define USE_ASYNC_LDS 1
#endif
#endif
#ifndef USE_ASYNC_LDS
#define USE_ASYNC_LDS 0
#endif

typedef __attribute__((ext_vector_type(16))) __bf16 v16bf;
typedef __attribute__((ext_vector_type(8)))  float  v8f;
typedef __attribute__((ext_vector_type(4)))  int    i32x4;
typedef __attribute__((address_space(1))) i32x4 glb_i32x4;   // global 16B chunk
typedef __attribute__((address_space(3))) i32x4 lds_i32x4;   // LDS 16B chunk

union BFrag { v16bf v; unsigned short u[16]; };

static __device__ __forceinline__ unsigned short f2bf(float f) {
    unsigned int u = __float_as_uint(f);
    unsigned int r = u + 0x7FFFu + ((u >> 16) & 1u);   // round-to-nearest-even
    return (unsigned short)(r >> 16);
}

// ---------------------------------------------------------------------------
// Stage 1: scale = M @ Ws^T, softmax over t (axis=1) per (b,s) column,
// store P[b][s][t] = softmax(scale)[b,t,s]  (so edge gather is P[b,src,dst]).
// ---------------------------------------------------------------------------
__global__ __launch_bounds__(128)
void attn_softmax_kernel(const float* __restrict__ M, const float* __restrict__ Ws,
                         float* __restrict__ P) {
    int bs = blockIdx.x;
    int b = bs / TT, s = bs - b * TT;
    __shared__ float wrow[DD];
    __shared__ float sc[TT];
    __shared__ float red[2];
    for (int d = threadIdx.x; d < DD; d += blockDim.x) wrow[d] = Ws[s * DD + d];
    __syncthreads();
    int t = threadIdx.x;
    if (t < TT) {
        const float* mrow = M + ((size_t)b * TT + t) * DD;
        float acc = 0.f;
        #pragma unroll 4
        for (int d = 0; d < DD; ++d) acc += mrow[d] * wrow[d];
        sc[t] = acc;
    }
    __syncthreads();
    if (threadIdx.x == 0) {
        float mx = -1e30f;
        for (int i = 0; i < TT; ++i) mx = fmaxf(mx, sc[i]);
        float sum = 0.f;
        for (int i = 0; i < TT; ++i) sum += __expf(sc[i] - mx);
        red[0] = mx; red[1] = 1.f / sum;
    }
    __syncthreads();
    if (t < TT)
        P[((size_t)b * TT + s) * TT + t] = __expf(sc[t] - red[0]) * red[1];
}

// ---------------------------------------------------------------------------
// Stage 2: W[r] = sum_n comp[r,n]*bases[n], stored bf16 column-major:
//   Wt[r][h][d]. Also root stored bf16 column-major: RootT[h][d].
// ---------------------------------------------------------------------------
__global__ __launch_bounds__(256)
void build_weights_kernel(const float* __restrict__ bases, const float* __restrict__ comp,
                          const float* __restrict__ root,
                          unsigned short* __restrict__ Wt, unsigned short* __restrict__ RootT) {
    const int DH = DD * HH;
    int idx = blockIdx.x * blockDim.x + threadIdx.x;
    if (idx < RR * DH) {
        int r = idx / DH;
        int rem = idx - r * DH;
        int h = rem >> 8, d = rem & 255;
        float acc = 0.f;
        #pragma unroll
        for (int n = 0; n < NB; ++n)
            acc += comp[r * NB + n] * bases[((size_t)n * DD + d) * HH + h];
        Wt[idx] = f2bf(acc);                         // [r][h][d]
    } else if (idx < (RR + 1) * DH) {
        int rem = idx - RR * DH;
        int h = rem >> 8, d = rem & 255;
        RootT[rem] = f2bf(root[(size_t)d * HH + h]); // [h][d]
    }
}

// ---------------------------------------------------------------------------
// Stage 3: x (fp32) -> bf16
// ---------------------------------------------------------------------------
__global__ __launch_bounds__(256)
void convert_x_kernel(const float* __restrict__ x, unsigned short* __restrict__ Xb) {
    size_t i = (size_t)blockIdx.x * blockDim.x + threadIdx.x;
    if (i < (size_t)NN * DD) Xb[i] = f2bf(x[i]);
}

// ---------------------------------------------------------------------------
// WMMA fragment layouts used here (wave32, A and B share contiguous K order):
// A 16x32 bf16:  lane holds row (lane&15); elems 0..15 -> K = k0 + 16*(lane>>4)+p
// B 32x16 bf16 (stored col-major [n][k]): lane -> col (lane&15),
//                16 contiguous K at k0 + 16*(lane>>4)  (one aligned 32B load)
// D f32: VGPR v -> row v + 8*(lane>>4), col lane&15
// ---------------------------------------------------------------------------

// Load a 16x256 bf16 A-tile into LDS with per-row gather g(row).
// Async path: global_load_async_to_lds_b128 (ASYNCcnt) — LDS written directly,
// no VGPR round-trip. Fallback: global load + ds store.
template <typename RowMap>
static __device__ __forceinline__ void load_a_tile(unsigned short* As,
                                                   const unsigned short* __restrict__ Xb,
                                                   RowMap g) {
#if USE_ASYNC_LDS
    {
        __attribute__((address_space(3))) unsigned short* AsL =
            (__attribute__((address_space(3))) unsigned short*)As;
        #pragma unroll
        for (int c = threadIdx.x; c < 512; c += 256) {       // 512 x 16B chunks
            int row = c >> 5, kc = c & 31;                   // kc: 16B chunk in row
            unsigned short* src = const_cast<unsigned short*>(Xb) + (size_t)g(row) * DD + kc * 8;
            __builtin_amdgcn_global_load_async_to_lds_b128(
                (glb_i32x4*)src, (lds_i32x4*)(AsL + c * 8), 0, 0);
        }
        asm volatile("s_wait_asynccnt 0" ::: "memory");
    }
    __syncthreads();
#else
    for (int i = threadIdx.x; i < 16 * DD; i += 256) {
        int row = i >> 8, k = i & 255;
        As[i] = Xb[(size_t)g(row) * DD + k];
    }
    __syncthreads();
#endif
}

// Shared 16xK-tile GEMM body: wave computes two 16x16 N-tiles over K=256.
static __device__ __forceinline__ void gemm_tile(const unsigned short* As,
                                                 const unsigned short* __restrict__ Bmat,
                                                 v8f& acc0, v8f& acc1,
                                                 int hv, int lrow, int n0) {
    #pragma unroll
    for (int kb = 0; kb < 8; ++kb) {
        int k0 = kb * 32;
        BFrag a;
        #pragma unroll
        for (int p = 0; p < 16; ++p) a.u[p] = As[lrow * DD + k0 + hv * 16 + p];
        v16bf b0 = *(const v16bf*)(Bmat + ((size_t)(n0 * 16 + lrow)) * DD + k0 + hv * 16);
        v16bf b1 = *(const v16bf*)(Bmat + ((size_t)((n0 + 1) * 16 + lrow)) * DD + k0 + hv * 16);
        acc0 = __builtin_amdgcn_wmma_f32_16x16x32_bf16(false, a.v, false, b0, (short)0, acc0, false, false);
        acc1 = __builtin_amdgcn_wmma_f32_16x16x32_bf16(false, a.v, false, b1, (short)0, acc1, false, false);
    }
}

// Stage 4: Y[r][m][h] = Xsrc[m] @ W[r], m = b*TT + t (gathered rows of x).
__global__ __launch_bounds__(256)
void rgcn_transform_kernel(const unsigned short* __restrict__ Xb,
                           const unsigned short* __restrict__ Wt,
                           float* __restrict__ Y) {
    int tile = blockIdx.x;        // M tile: 0..359
    int r    = blockIdx.y;        // relation
    __shared__ unsigned short As[16 * DD];
    load_a_tile(As, Xb, [tile](int row) {
        int m = tile * 16 + row;
        int b = m / TT, t = m - b * TT;
        return b * NPG + t;                  // gather: only first TT nodes per graph
    });

    int wave = threadIdx.x >> 5;
    int lane = threadIdx.x & 31;
    int hv   = lane >> 4;
    int lrow = lane & 15;
    int n0 = wave * 2;
    v8f acc0 = {}; v8f acc1 = {};
    gemm_tile(As, Wt + (size_t)r * DD * HH, acc0, acc1, hv, lrow, n0);

    float* Yr = Y + ((size_t)r * MSRC + tile * 16) * HH;
    #pragma unroll
    for (int v = 0; v < 8; ++v) {
        int mrow = v + hv * 8;
        Yr[(size_t)mrow * HH + n0 * 16 + lrow]       = acc0[v];
        Yr[(size_t)mrow * HH + (n0 + 1) * 16 + lrow] = acc1[v];
    }
}

// Stage 5: Out = x @ root + bias  (initializes every output element)
__global__ __launch_bounds__(256)
void root_transform_kernel(const unsigned short* __restrict__ Xb,
                           const unsigned short* __restrict__ RootT,
                           const float* __restrict__ bias,
                           float* __restrict__ Out) {
    int tile = blockIdx.x;        // 0..379
    __shared__ unsigned short As[16 * DD];
    load_a_tile(As, Xb, [tile](int row) { return tile * 16 + row; });

    int wave = threadIdx.x >> 5;
    int lane = threadIdx.x & 31;
    int hv   = lane >> 4;
    int lrow = lane & 15;
    int n0 = wave * 2;
    v8f acc0 = {}; v8f acc1 = {};
    gemm_tile(As, RootT, acc0, acc1, hv, lrow, n0);

    float* Orow = Out + (size_t)tile * 16 * HH;
    #pragma unroll
    for (int v = 0; v < 8; ++v) {
        int mrow = v + hv * 8;
        int c0 = n0 * 16 + lrow, c1 = (n0 + 1) * 16 + lrow;
        Orow[(size_t)mrow * HH + c0] = acc0[v] + bias[c0];
        Orow[(size_t)mrow * HH + c1] = acc1[v] + bias[c1];
    }
}

// ---------------------------------------------------------------------------
// Stage 6: per-edge scatter: Out[dst] += norm * Y[type][b*TT+src].
// One block per edge, thread = h channel. Y (29.5MB) lives in 192MB L2.
// ---------------------------------------------------------------------------
__global__ __launch_bounds__(256)
void edge_scatter_kernel(const int* __restrict__ esrc, const int* __restrict__ edst,
                         const int* __restrict__ etype,
                         const float* __restrict__ node_att, const float* __restrict__ P,
                         const float* __restrict__ Y, float* __restrict__ Out) {
    int e = blockIdx.x;                 // 0 .. ETOT-1
    int b = e / EPB;
    int j = e - b * EPB;
    int src, dst, type;
    float norm;
    if (j < EA) {
        src  = esrc[(size_t)b * EA + j];
        dst  = edst[(size_t)b * EA + j];
        type = etype[(size_t)b * EA + j];
        norm = P[((size_t)b * TT + src) * TT + dst];
    } else {
        int k = j - EA;
        int t = k / OO, o = k - t * OO;
        src = t; dst = TT + o; type = RR - 1;
        norm = node_att[((size_t)b * TT + t) * OO + o];
    }
    int h = threadIdx.x;
    float v = norm * Y[((size_t)type * MSRC + (size_t)b * TT + src) * HH + h];
    unsafeAtomicAdd(&Out[((size_t)b * NPG + dst) * HH + h], v);
}

// ---------------------------------------------------------------------------
extern "C" void kernel_launch(void* const* d_in, const int* in_sizes, int n_in,
                              void* d_out, int out_size, void* d_ws, size_t ws_size,
                              hipStream_t stream) {
    const float* M        = (const float*)d_in[0];
    const float* x        = (const float*)d_in[1];
    const float* node_att = (const float*)d_in[2];
    const float* Ws       = (const float*)d_in[3];
    const float* bases    = (const float*)d_in[4];
    const float* comp     = (const float*)d_in[5];
    const float* root     = (const float*)d_in[6];
    const float* bias     = (const float*)d_in[7];
    const int*   esrc     = (const int*)d_in[8];
    const int*   edst     = (const int*)d_in[9];
    const int*   etype    = (const int*)d_in[10];
    float* Out = (float*)d_out;

    // Workspace carve (256B aligned regions)
    char* w = (char*)d_ws;
    size_t off = 0;
    auto carve = [&](size_t bytes) -> void* {
        void* p = w + off;
        off = (off + bytes + 255) & ~(size_t)255;
        return p;
    };
    float*          P     = (float*)carve((size_t)BB * TT * TT * sizeof(float));         // 2.07 MB
    float*          Y     = (float*)carve((size_t)RR * MSRC * HH * sizeof(float));       // 29.5 MB
    unsigned short* Xb    = (unsigned short*)carve((size_t)NN * DD * 2);                 // 3.11 MB
    unsigned short* Wt    = (unsigned short*)carve((size_t)RR * DD * HH * 2);            // 655 KB
    unsigned short* RootT = (unsigned short*)carve((size_t)DD * HH * 2);                 // 131 KB
    (void)ws_size; (void)in_sizes; (void)n_in; (void)out_size;

    // 1. attention softmax -> P[b][s][t]
    attn_softmax_kernel<<<BB * TT, 128, 0, stream>>>(M, Ws, P);
    // 2. basis-combined weights (bf16, col-major) + root (bf16, col-major)
    build_weights_kernel<<<((RR + 1) * DD * HH) / 256, 256, 0, stream>>>(bases, comp, root, Wt, RootT);
    // 3. x -> bf16
    convert_x_kernel<<<(NN * DD) / 256, 256, 0, stream>>>(x, Xb);
    // 4. Y[r] = Xsrc @ W[r]   (WMMA bf16, A tile staged via async-to-LDS)
    rgcn_transform_kernel<<<dim3(MSRC / 16, RR), 256, 0, stream>>>(Xb, Wt, Y);
    // 5. Out = x @ root + bias (WMMA bf16; initializes d_out)
    root_transform_kernel<<<NN / 16, 256, 0, stream>>>(Xb, RootT, bias, Out);
    // 6. edge scatter with fp32 hardware atomics
    edge_scatter_kernel<<<ETOT, 256, 0, stream>>>(esrc, edst, etype, node_att, P, Y, Out);
}
